// DecLayer_74637941670411
// MI455X (gfx1250) — matrix operand order repaired and checked by reading, not compiled, not run on hardware
//
#include <hip/hip_runtime.h>
#include <hip/hip_bf16.h>
#include <math.h>

#define Hdim 128
#define Cdim 384
#define Kn   32
#define HC   512
#define SCALE 30.0f
#define LNEPS 1e-5f

typedef __attribute__((ext_vector_type(16))) _Float16 v16h;
typedef __attribute__((ext_vector_type(8)))  _Float16 v8h;
typedef __attribute__((ext_vector_type(4)))  _Float16 v4h;
typedef __attribute__((ext_vector_type(8)))  float    v8f;

__device__ __forceinline__ float gelu_f(float x) {
    // exact erf GELU to match torch/jax approximate=False
    return 0.5f * x * (1.0f + erff(x * 0.7071067811865475f));
}

__device__ __forceinline__ v16h cat8(v8h lo, v8h hi) {
    return __builtin_shufflevector(lo, hi, 0,1,2,3,4,5,6,7,8,9,10,11,12,13,14,15);
}

__device__ __forceinline__ v8f wmma_f16(v16h a, v16h b, v8f c) {
    return __builtin_amdgcn_wmma_f32_16x16x32_f16(false, a, false, b, (short)0, c, false, false);
}

// ---------------------------------------------------------------------------
// Weight prep: dst[o][i] (f16, [out][in] row-major) = (f16) src[i][o]
// Makes every WMMA B-fragment two contiguous 16B runs per lane.
// ---------------------------------------------------------------------------
__global__ __launch_bounds__(256) void wcvt_kernel(const float* __restrict__ src,
                                                   _Float16* __restrict__ dst,
                                                   int in_dim, int out_dim) {
    int idx = blockIdx.x * 256 + threadIdx.x;
    int total = in_dim * out_dim;
    if (idx >= total) return;
    int o = idx / in_dim;
    int i = idx - o * in_dim;
    dst[(size_t)o * in_dim + i] = (_Float16)src[(size_t)i * out_dim + o];
}

// ---------------------------------------------------------------------------
// Message kernel: one workgroup (256 thr = 8 waves) per node.
// dh[n][:] = sum_k mask_attend[n][k] * MLP3(concat(h_V[n], h_E[n][k])) / 30
// ---------------------------------------------------------------------------
__global__ __launch_bounds__(256) void msg_kernel(
    const float* __restrict__ hV, const float* __restrict__ hE,
    const float* __restrict__ maskA,
    const _Float16* __restrict__ W1T, const float* __restrict__ b1,
    const _Float16* __restrict__ W2T, const float* __restrict__ b2,
    const _Float16* __restrict__ W3T, const float* __restrict__ b3,
    float* __restrict__ dh, int N)
{
    __shared__ _Float16 sA[Kn][HC];     // h_EV in f16, 32 KB
    __shared__ _Float16 sH0[Kn][Hdim];  // 8 KB
    __shared__ _Float16 sH1[Kn][Hdim];  // 8 KB
    __shared__ float    smask[Kn];

    const int tid  = threadIdx.x;
    const int lane = tid & 31;
    const int wave = tid >> 5;      // 0..7 -> N-tile
    const int hi   = lane >> 4;     // half-wave select
    const int l15  = lane & 15;

    for (int n = blockIdx.x; n < N; n += gridDim.x) {
        // ---- Phase A: stage h_EV = [h_V | h_E] as f16 in LDS ----
        const float* hvn = hV + (size_t)n * Hdim;
        for (int idx = tid; idx < Kn * Hdim; idx += 256) {
            int r = idx >> 7, c = idx & (Hdim - 1);
            sA[r][c] = (_Float16)hvn[c];
        }
        const float4* hen = (const float4*)(hE + (size_t)n * Kn * Cdim);
        for (int idx = tid; idx < Kn * (Cdim / 4); idx += 256) {
            float4 v = hen[idx];
            int r = idx / (Cdim / 4), c4 = idx % (Cdim / 4);
            v4h h = { (_Float16)v.x, (_Float16)v.y, (_Float16)v.z, (_Float16)v.w };
            *(v4h*)&sA[r][Hdim + c4 * 4] = h;
        }
        if (tid < Kn) smask[tid] = maskA[(size_t)n * Kn + tid];
        __syncthreads();

        // ---- GEMM1: [32x512] x [512x128] -> gelu -> sH0 ----
        {
            v8f c0 = {}, c1 = {};
            const _Float16* brow = W1T + (size_t)(wave * 16 + l15) * HC + hi * 8;
            #pragma unroll 4
            for (int kt = 0; kt < 16; ++kt) {
                int kb = kt * 32;
                v16h a0 = cat8(*(const v8h*)&sA[l15     ][kb + hi * 8],
                               *(const v8h*)&sA[l15     ][kb + hi * 8 + 16]);
                v16h a1 = cat8(*(const v8h*)&sA[16 + l15][kb + hi * 8],
                               *(const v8h*)&sA[16 + l15][kb + hi * 8 + 16]);
                v16h b  = cat8(*(const v8h*)&brow[kb], *(const v8h*)&brow[kb + 16]);
                c0 = wmma_f16(a0, b, c0);
                c1 = wmma_f16(a1, b, c1);
            }
            float bias = b1[wave * 16 + l15];
            #pragma unroll
            for (int r = 0; r < 8; ++r) {
                sH0[r + hi * 8     ][wave * 16 + l15] = (_Float16)gelu_f(c0[r] + bias);
                sH0[16 + r + hi * 8][wave * 16 + l15] = (_Float16)gelu_f(c1[r] + bias);
            }
        }
        __syncthreads();

        // ---- GEMM2: [32x128] x [128x128] -> gelu -> sH1 ----
        {
            v8f c0 = {}, c1 = {};
            const _Float16* brow = W2T + (size_t)(wave * 16 + l15) * Hdim + hi * 8;
            #pragma unroll
            for (int kt = 0; kt < 4; ++kt) {
                int kb = kt * 32;
                v16h a0 = cat8(*(const v8h*)&sH0[l15     ][kb + hi * 8],
                               *(const v8h*)&sH0[l15     ][kb + hi * 8 + 16]);
                v16h a1 = cat8(*(const v8h*)&sH0[16 + l15][kb + hi * 8],
                               *(const v8h*)&sH0[16 + l15][kb + hi * 8 + 16]);
                v16h b  = cat8(*(const v8h*)&brow[kb], *(const v8h*)&brow[kb + 16]);
                c0 = wmma_f16(a0, b, c0);
                c1 = wmma_f16(a1, b, c1);
            }
            float bias = b2[wave * 16 + l15];
            #pragma unroll
            for (int r = 0; r < 8; ++r) {
                sH1[r + hi * 8     ][wave * 16 + l15] = (_Float16)gelu_f(c0[r] + bias);
                sH1[16 + r + hi * 8][wave * 16 + l15] = (_Float16)gelu_f(c1[r] + bias);
            }
        }
        __syncthreads();

        // ---- GEMM3 + masked neighbor-sum reduce ----
        {
            v8f c0 = {}, c1 = {};
            const _Float16* brow = W3T + (size_t)(wave * 16 + l15) * Hdim + hi * 8;
            #pragma unroll
            for (int kt = 0; kt < 4; ++kt) {
                int kb = kt * 32;
                v16h a0 = cat8(*(const v8h*)&sH1[l15     ][kb + hi * 8],
                               *(const v8h*)&sH1[l15     ][kb + hi * 8 + 16]);
                v16h a1 = cat8(*(const v8h*)&sH1[16 + l15][kb + hi * 8],
                               *(const v8h*)&sH1[16 + l15][kb + hi * 8 + 16]);
                v16h b  = cat8(*(const v8h*)&brow[kb], *(const v8h*)&brow[kb + 16]);
                c0 = wmma_f16(a0, b, c0);
                c1 = wmma_f16(a1, b, c1);
            }
            float bias = b3[wave * 16 + l15];
            float s = 0.f;
            #pragma unroll
            for (int r = 0; r < 8; ++r) {
                s += (c0[r] + bias) * smask[r + hi * 8];        // rows 0..15 half
                s += (c1[r] + bias) * smask[16 + r + hi * 8];   // rows 16..31 half
            }
            s += __shfl_xor(s, 16, 32);   // lane l and l^16 hold complementary rows
            if (hi == 0)
                dh[(size_t)n * Hdim + wave * 16 + l15] = s * (1.0f / SCALE);
        }
        __syncthreads();   // protect sA/smask before next iteration
    }
}

// ---------------------------------------------------------------------------
// Node kernel: 32 nodes per workgroup. LN1 -> FFN (WMMA) -> residual -> LN2.
// ---------------------------------------------------------------------------
__global__ __launch_bounds__(256) void node_kernel(
    const float* __restrict__ hV, const float* __restrict__ dh,
    const float* __restrict__ maskV,
    const float* __restrict__ g1, const float* __restrict__ bb1,
    const _Float16* __restrict__ WinT, const float* __restrict__ bin,
    const _Float16* __restrict__ WoutT, const float* __restrict__ bout,
    const float* __restrict__ g2, const float* __restrict__ bb2,
    float* __restrict__ out, int N)
{
    __shared__ float    xf[32][Hdim];   // LN1 output, fp32 (for residual)  16 KB
    __shared__ _Float16 xh[32][Hdim];   // LN1 output, f16 (GEMM A)          8 KB
    __shared__ _Float16 sT[32][HC];     // gelu(x@Win+b), f16               32 KB
    __shared__ float    yf[32][Hdim];   // FFN out + residual               16 KB

    const int tid  = threadIdx.x;
    const int lane = tid & 31;
    const int wave = tid >> 5;
    const int hi   = lane >> 4;
    const int l15  = lane & 15;
    const int n0   = blockIdx.x * 32;

    // ---- LN1: x = LN(h_V + dh); 8 lanes per row ----
    {
        int row = tid >> 3, j = tid & 7;
        int gn = n0 + row;
        float v[16];
        float s = 0.f, ss = 0.f;
        #pragma unroll
        for (int c = 0; c < 16; ++c) {
            int col = j * 16 + c;
            float t = (gn < N) ? (hV[(size_t)gn * Hdim + col] + dh[(size_t)gn * Hdim + col]) : 0.f;
            v[c] = t; s += t; ss += t * t;
        }
        s  += __shfl_xor(s, 1, 32);  s  += __shfl_xor(s, 2, 32);  s  += __shfl_xor(s, 4, 32);
        ss += __shfl_xor(ss, 1, 32); ss += __shfl_xor(ss, 2, 32); ss += __shfl_xor(ss, 4, 32);
        float mean = s * (1.0f / Hdim);
        float var  = ss * (1.0f / Hdim) - mean * mean;
        float inv  = rsqrtf(var + LNEPS);
        #pragma unroll
        for (int c = 0; c < 16; ++c) {
            int col = j * 16 + c;
            float xn = (v[c] - mean) * inv * g1[col] + bb1[col];
            xf[row][col] = xn;
            xh[row][col] = (_Float16)xn;
        }
    }
    __syncthreads();

    // ---- GEMM A: [32x128] x [128x512] -> gelu -> sT ----
    {
        v8f acc0[4] = {}, acc1[4] = {};
        #pragma unroll
        for (int kt = 0; kt < 4; ++kt) {
            int kb = kt * 32;
            v16h a0 = cat8(*(const v8h*)&xh[l15     ][kb + hi * 8],
                           *(const v8h*)&xh[l15     ][kb + hi * 8 + 16]);
            v16h a1 = cat8(*(const v8h*)&xh[16 + l15][kb + hi * 8],
                           *(const v8h*)&xh[16 + l15][kb + hi * 8 + 16]);
            #pragma unroll
            for (int j = 0; j < 4; ++j) {
                int ncol = wave * 64 + j * 16 + l15;
                v16h b = cat8(*(const v8h*)&WinT[(size_t)ncol * Hdim + kb + hi * 8], *(const v8h*)&WinT[(size_t)ncol * Hdim + kb + hi * 8 + 16]);
                acc0[j] = wmma_f16(a0, b, acc0[j]);
                acc1[j] = wmma_f16(a1, b, acc1[j]);
            }
        }
        #pragma unroll
        for (int j = 0; j < 4; ++j) {
            int ncol = wave * 64 + j * 16 + l15;
            float bias = bin[ncol];
            #pragma unroll
            for (int r = 0; r < 8; ++r) {
                sT[r + hi * 8     ][ncol] = (_Float16)gelu_f(acc0[j][r] + bias);
                sT[16 + r + hi * 8][ncol] = (_Float16)gelu_f(acc1[j][r] + bias);
            }
        }
    }
    __syncthreads();

    // ---- GEMM B: [32x512] x [512x128] -> + bias + residual -> yf ----
    {
        v8f c0 = {}, c1 = {};
        const _Float16* brow = WoutT + (size_t)(wave * 16 + l15) * HC + hi * 8;
        #pragma unroll 4
        for (int kt = 0; kt < 16; ++kt) {
            int kb = kt * 32;
            v16h a0 = cat8(*(const v8h*)&sT[l15     ][kb + hi * 8],
                           *(const v8h*)&sT[l15     ][kb + hi * 8 + 16]);
            v16h a1 = cat8(*(const v8h*)&sT[16 + l15][kb + hi * 8],
                           *(const v8h*)&sT[16 + l15][kb + hi * 8 + 16]);
            v16h b  = cat8(*(const v8h*)&brow[kb], *(const v8h*)&brow[kb + 16]);
            c0 = wmma_f16(a0, b, c0);
            c1 = wmma_f16(a1, b, c1);
        }
        int ncol = wave * 16 + l15;
        float bias = bout[ncol];
        #pragma unroll
        for (int r = 0; r < 8; ++r) {
            yf[r + hi * 8     ][ncol] = c0[r] + bias + xf[r + hi * 8     ][ncol];
            yf[16 + r + hi * 8][ncol] = c1[r] + bias + xf[16 + r + hi * 8][ncol];
        }
    }
    __syncthreads();

    // ---- LN2 + mask_V + store ----
    {
        int row = tid >> 3, j = tid & 7;
        int gn = n0 + row;
        float v[16];
        float s = 0.f, ss = 0.f;
        #pragma unroll
        for (int c = 0; c < 16; ++c) {
            float t = yf[row][j * 16 + c];
            v[c] = t; s += t; ss += t * t;
        }
        s  += __shfl_xor(s, 1, 32);  s  += __shfl_xor(s, 2, 32);  s  += __shfl_xor(s, 4, 32);
        ss += __shfl_xor(ss, 1, 32); ss += __shfl_xor(ss, 2, 32); ss += __shfl_xor(ss, 4, 32);
        float mean = s * (1.0f / Hdim);
        float var  = ss * (1.0f / Hdim) - mean * mean;
        float inv  = rsqrtf(var + LNEPS);
        if (gn < N) {
            float mk = maskV[gn];
            #pragma unroll
            for (int c = 0; c < 16; ++c) {
                int col = j * 16 + c;
                out[(size_t)gn * Hdim + col] = mk * ((v[c] - mean) * inv * g2[col] + bb2[col]);
            }
        }
    }
}

// ---------------------------------------------------------------------------
extern "C" void kernel_launch(void* const* d_in, const int* in_sizes, int n_in,
                              void* d_out, int out_size, void* d_ws, size_t ws_size,
                              hipStream_t stream)
{
    (void)n_in; (void)out_size; (void)ws_size;
    const float* hV    = (const float*)d_in[0];
    const float* hE    = (const float*)d_in[1];
    const float* maskV = (const float*)d_in[2];
    const float* maskA = (const float*)d_in[3];
    const float* W1w   = (const float*)d_in[4];
    const float* W1b   = (const float*)d_in[5];
    const float* W2w   = (const float*)d_in[6];
    const float* W2b   = (const float*)d_in[7];
    const float* W3w   = (const float*)d_in[8];
    const float* W3b   = (const float*)d_in[9];
    const float* ln1g  = (const float*)d_in[10];
    const float* ln1b  = (const float*)d_in[11];
    const float* Winw  = (const float*)d_in[12];
    const float* Winb  = (const float*)d_in[13];
    const float* Woutw = (const float*)d_in[14];
    const float* Woutb = (const float*)d_in[15];
    const float* ln2g  = (const float*)d_in[16];
    const float* ln2b  = (const float*)d_in[17];
    float* out = (float*)d_out;
    int N = in_sizes[0] / Hdim;

    // workspace layout: f16 transposed weights, then fp32 dh buffer
    _Float16* W1T   = (_Float16*)d_ws;        // [128][512]
    _Float16* W2T   = W1T   + 512 * 128;      // [128][128]
    _Float16* W3T   = W2T   + 128 * 128;      // [128][128]
    _Float16* WinT  = W3T   + 128 * 128;      // [512][128]
    _Float16* WoutT = WinT  + 512 * 128;      // [128][512]
    float*    dhb   = (float*)(WoutT + 128 * 512);  // [N][128]

    auto cvt = [&](const float* src, _Float16* dst, int in_dim, int out_dim) {
        int total = in_dim * out_dim;
        wcvt_kernel<<<(total + 255) / 256, 256, 0, stream>>>(src, dst, in_dim, out_dim);
    };
    cvt(W1w,   W1T,   512, 128);
    cvt(W2w,   W2T,   128, 128);
    cvt(W3w,   W3T,   128, 128);
    cvt(Winw,  WinT,  128, 512);
    cvt(Woutw, WoutT, 512, 128);

    msg_kernel<<<N, 256, 0, stream>>>(hV, hE, maskA, W1T, W1b, W2T, W2b, W3T, W3b, dhb, N);
    node_kernel<<<(N + 31) / 32, 256, 0, stream>>>(hV, dhb, maskV, ln1g, ln1b,
                                                   WinT, Winb, WoutT, Woutb,
                                                   ln2g, ln2b, out, N);
}
